// Decoder_8375186227401
// MI455X (gfx1250) — matrix operand order repaired and checked by reading, not compiled, not run on hardware
//
#include <hip/hip_runtime.h>
#include <cstdint>

// Problem dims (fixed by reference)
#define T_ENC  512
#define NB     256
#define LSTEPS 250
#define KSZ    128
#define VSZ    128
#define HSZ    512
#define VOC    35
#define G1     (4*HSZ)          // 2048 gate outputs, LSTM1
#define K1     (HSZ+VSZ+HSZ)    // 1152 = [emb | ctx | h1]
#define G2     (4*KSZ)          // 512 gate outputs, LSTM2
#define K2     (HSZ+KSZ)        // 640  = [h1 | h2]

typedef __attribute__((ext_vector_type(16))) __bf16 v16bf;
typedef __attribute__((ext_vector_type(8)))  float  v8f;

__device__ __forceinline__ float sigmf(float x) { return 1.0f / (1.0f + expf(-x)); }

// ---- WMMA fragment loaders (wave32, contiguous k = 16*(lane/16) + elem) -------------------
// A: 16x32 bf16, row-major source, leading dim ld.
// lanes 0-15 : M=lane,   K=0..15
// lanes16-31 : M=lane-16,K=16..31
__device__ __forceinline__ v16bf load_frag_a(const __bf16* __restrict__ src, int ld) {
    const int lane = threadIdx.x & 31;
    const int m    = lane & 15;
    const int kb   = (lane >> 4) << 4;      // 0 or 16
    v16bf a;
#pragma unroll
    for (int v = 0; v < 8; ++v) {
        const int k = kb + (v << 1);
        a[2 * v]     = src[m * ld + k];
        a[2 * v + 1] = src[m * ld + k + 1];
    }
    return a;
}

// B fragment from an LDS tile staged as [n(16)][k(32)] bf16 (64B rows).
// lanes 0-15 : col=lane,    K=0..15   -> 32 contiguous bytes
// lanes16-31 : col=lane-16, K=16..31  -> 32 contiguous bytes
__device__ __forceinline__ v16bf load_frag_b_lds(const __bf16* base) {
    const int lane = threadIdx.x & 31;
    const __bf16* p = base + (lane & 15) * 32 + ((lane >> 4) << 4);
    return *(const v16bf*)p;
}

// Issue async staging of one k-slab of B (2 N-tiles x 16 rows x 32 k, bf16 = 2KB)
// into LDS at byte address ldsbase. Threads 0..127 issue one b128 each.
__device__ __forceinline__ void async_stage_b(const __bf16* __restrict__ W, int ld,
                                              int tn0, int kk, unsigned ldsbase) {
    const int t = threadIdx.x;
    if (t < 128) {
        const int tile = t >> 6;         // 0..1
        const int c    = t & 63;         // chunk within tile
        const int r    = c >> 2;         // row 0..15
        const int seg  = c & 3;          // 16B segment 0..3
        const __bf16* g = W + (size_t)((tn0 + tile) * 16 + r) * ld + kk + seg * 8;
        const unsigned l = ldsbase + tile * 1024 + r * 64 + seg * 16;
        asm volatile("global_load_async_to_lds_b128 %0, %1, off"
                     :: "v"(l), "v"(g) : "memory");
    }
}

__device__ __forceinline__ void wait_async_all() {
    asm volatile("s_wait_asynccnt 0x0" ::: "memory");
}

// ---- GEMM: C(MxNOUT,f32) = A(MxK,bf16) * W(NOUTxK,bf16)^T + bias ----------
// Block = 8 waves: covers 8 M-tiles x 2 N-tiles. B slab staged to LDS with
// async-to-LDS double buffering; A frag reused across the 2 N-tiles.
template <int M, int NOUT, int K>
__global__ void gemm_bf16_wmma(const __bf16* __restrict__ A,
                               const __bf16* __restrict__ W,
                               const float* __restrict__ bias,
                               float* __restrict__ C) {
    __shared__ __attribute__((aligned(64))) __bf16 sB[2][2 * 16 * 32]; // 2 bufs x 2KB
    const int wave = threadIdx.x >> 5;
    const int NBM  = (M / 16) / 8;                 // M-blocks per column
    const int bM   = blockIdx.x % NBM;
    const int bN   = blockIdx.x / NBM;             // N tile-pair
    const int tm   = bM * 8 + wave;
    const int tn0  = bN * 2;

    const unsigned lds0 = (unsigned)(unsigned long long)(void*)&sB[0][0];
    const unsigned lds1 = (unsigned)(unsigned long long)(void*)&sB[1][0];

    const __bf16* a0 = A + (size_t)tm * 16 * K;

    async_stage_b(W, K, tn0, 0, lds0);             // prologue: k=0 -> buf0

    v8f acc0 = {0.f, 0.f, 0.f, 0.f, 0.f, 0.f, 0.f, 0.f};
    v8f acc1 = {0.f, 0.f, 0.f, 0.f, 0.f, 0.f, 0.f, 0.f};
    int buf = 0;
    for (int kk = 0; kk < K; kk += 32) {
        wait_async_all();                          // this wave's async stores done
        __syncthreads();                           // -> whole slab visible to block
        if (kk + 32 < K)
            async_stage_b(W, K, tn0, kk + 32, buf ? lds0 : lds1);
        __builtin_prefetch(a0 + kk + 32, 0, 1);    // global_prefetch_b8
        v16bf a  = load_frag_a(a0 + kk, K);
        const __bf16* sb = &sB[buf][0];
        v16bf b0 = load_frag_b_lds(sb);
        v16bf b1 = load_frag_b_lds(sb + 512);
        acc0 = __builtin_amdgcn_wmma_f32_16x16x32_bf16(false, a, false, b0,
                                                       (short)0, acc0, false, false);
        acc1 = __builtin_amdgcn_wmma_f32_16x16x32_bf16(false, a, false, b1,
                                                       (short)0, acc1, false, false);
        __syncthreads();                           // block done reading buf
        buf ^= 1;
    }

    const int lane  = threadIdx.x & 31;
    const int rbase = tm * 16 + ((lane >> 4) << 3);
    {
        const int col  = tn0 * 16 + (lane & 15);
        const float bb = bias[col];
#pragma unroll
        for (int r = 0; r < 8; ++r)
            C[(size_t)(rbase + r) * NOUT + col] = acc0[r] + bb;
    }
    {
        const int col  = (tn0 + 1) * 16 + (lane & 15);
        const float bb = bias[col];
#pragma unroll
        for (int r = 0; r < 8; ++r)
            C[(size_t)(rbase + r) * NOUT + col] = acc1[r] + bb;
    }
}

// ---- one-time prep --------------------------------------------------------
__global__ void prep_w1cat(const float* __restrict__ w_ih1, const float* __restrict__ w_hh1,
                           __bf16* __restrict__ W1) {
    int idx = blockIdx.x * blockDim.x + threadIdx.x;
    if (idx >= G1 * K1) return;
    int g = idx / K1, k = idx % K1;
    float v = (k < (HSZ + VSZ)) ? w_ih1[g * (HSZ + VSZ) + k] : w_hh1[g * HSZ + (k - HSZ - VSZ)];
    W1[idx] = (__bf16)v;
}

__global__ void prep_w2cat(const float* __restrict__ w_ih2, const float* __restrict__ w_hh2,
                           __bf16* __restrict__ W2) {
    int idx = blockIdx.x * blockDim.x + threadIdx.x;
    if (idx >= G2 * K2) return;
    int g = idx / K2, k = idx % K2;
    float v = (k < HSZ) ? w_ih2[g * HSZ + k] : w_hh2[g * KSZ + (k - HSZ)];
    W2[idx] = (__bf16)v;
}

__global__ void prep_bias(const float* __restrict__ b_ih1, const float* __restrict__ b_hh1,
                          const float* __restrict__ b_ih2, const float* __restrict__ b_hh2,
                          float* __restrict__ b1, float* __restrict__ b2) {
    int idx = blockIdx.x * blockDim.x + threadIdx.x;
    if (idx < G1) b1[idx] = b_ih1[idx] + b_hh1[idx];
    if (idx < G2) b2[idx] = b_ih2[idx] + b_hh2[idx];
}

// (T,N,128) f32  ->  (N,T,128) bf16 ; same for values. Grid = T*N blocks x128.
__global__ void transpose_enc(const float* __restrict__ ek, const float* __restrict__ ev,
                              __bf16* __restrict__ keyT, __bf16* __restrict__ valT) {
    const int b = blockIdx.x;          // b = t*NB + n
    const int t = b >> 8;
    const int n = b & (NB - 1);
    const int k = threadIdx.x;
    const size_t src = (size_t)b * 128 + k;
    const size_t dst = ((size_t)n * T_ENC + t) * 128 + k;
    keyT[dst] = (__bf16)ek[src];
    valT[dst] = (__bf16)ev[src];
}

__global__ void init_state(const float* __restrict__ enc_val,
                           float* h1, float* c1, float* h2, float* c2, float* ctx) {
    int idx = blockIdx.x * blockDim.x + threadIdx.x;
    if (idx < NB * HSZ) { h1[idx] = 0.f; c1[idx] = 0.f; }
    if (idx < NB * KSZ) { h2[idx] = 0.f; c2[idx] = 0.f; ctx[idx] = enc_val[idx]; }  // enc_values[0]
}

// ---- per-step kernels -----------------------------------------------------
__global__ void build_x1(const int* __restrict__ text, const float* __restrict__ emb,
                         const float* __restrict__ ctx, const float* __restrict__ h1,
                         __bf16* __restrict__ xcat, int l) {
    const int n   = blockIdx.x;
    const int tok = text[l * NB + n];
    for (int k = threadIdx.x; k < K1; k += blockDim.x) {
        float v;
        if (k < HSZ)            v = (tok == 0) ? 0.f : emb[(size_t)tok * HSZ + k];
        else if (k < HSZ + VSZ) v = ctx[n * VSZ + (k - HSZ)];
        else                    v = h1[n * HSZ + (k - HSZ - VSZ)];
        xcat[(size_t)n * K1 + k] = (__bf16)v;
    }
}

__global__ void lstm_act1(const float* __restrict__ g1, float* __restrict__ c1,
                          float* __restrict__ h1, const float* __restrict__ h2,
                          __bf16* __restrict__ x2cat) {
    const int idx = blockIdx.x * blockDim.x + threadIdx.x;   // NB*HSZ threads
    const int n = idx >> 9, j = idx & (HSZ - 1);
    const float i = sigmf(g1[(size_t)n * G1 + j]);
    const float f = sigmf(g1[(size_t)n * G1 + j + HSZ]);
    const float g = tanhf(g1[(size_t)n * G1 + j + 2 * HSZ]);
    const float o = sigmf(g1[(size_t)n * G1 + j + 3 * HSZ]);
    const float c = f * c1[idx] + i * g;
    c1[idx] = c;
    const float h = o * tanhf(c);
    h1[idx] = h;
    x2cat[(size_t)n * K2 + j] = (__bf16)h;
    if (j < KSZ) x2cat[(size_t)n * K2 + HSZ + j] = (__bf16)h2[n * KSZ + j];
}

// Fused: LSTM2 pointwise + masked-softmax attention + output projection.
__global__ void attn_fused(const float* __restrict__ g2, float* __restrict__ c2,
                           float* __restrict__ h2,
                           const __bf16* __restrict__ keyT, const __bf16* __restrict__ valT,
                           const int* __restrict__ lens,
                           const float* __restrict__ w_out, const float* __restrict__ b_out,
                           float* __restrict__ ctx, float* __restrict__ out, int l) {
    __shared__ float sE[T_ENC];
    __shared__ float sh2[KSZ];
    __shared__ float sctx[KSZ];
    __shared__ float sred[256];
    const int n = blockIdx.x, tid = threadIdx.x;

    if (tid < KSZ) {
        const int j  = tid;
        const float i  = sigmf(g2[(size_t)n * G2 + j]);
        const float f  = sigmf(g2[(size_t)n * G2 + j + KSZ]);
        const float gg = tanhf(g2[(size_t)n * G2 + j + 2 * KSZ]);
        const float o  = sigmf(g2[(size_t)n * G2 + j + 3 * KSZ]);
        const float c  = f * c2[n * KSZ + j] + i * gg;
        c2[n * KSZ + j] = c;
        const float h = o * tanhf(c);
        h2[n * KSZ + j] = h;
        sh2[j] = h;
    }
    __syncthreads();

    const int len = lens[n];
    for (int t = tid; t < T_ENC; t += 256) {
        const __bf16* kp = keyT + ((size_t)n * T_ENC + t) * KSZ;
        float e = 0.f;
#pragma unroll 8
        for (int k = 0; k < KSZ; ++k) e += sh2[k] * (float)kp[k];
        sE[t] = (t < len) ? e : -1e9f;
    }
    __syncthreads();

    sred[tid] = fmaxf(sE[tid], sE[tid + 256]);
    for (int s = 128; s > 0; s >>= 1) { __syncthreads(); if (tid < s) sred[tid] = fmaxf(sred[tid], sred[tid + s]); }
    __syncthreads();
    const float mx = sred[0];
    __syncthreads();
    const float e0 = expf(sE[tid] - mx), e1 = expf(sE[tid + 256] - mx);
    sE[tid] = e0; sE[tid + 256] = e1;
    sred[tid] = e0 + e1;
    for (int s = 128; s > 0; s >>= 1) { __syncthreads(); if (tid < s) sred[tid] += sred[tid + s]; }
    __syncthreads();
    const float inv = 1.0f / sred[0];
    __syncthreads();

    const int col = tid & 127, half = tid >> 7;
    const __bf16* vp = valT + ((size_t)n * T_ENC + half * 256) * VSZ + col;
    float acc = 0.f;
#pragma unroll 4
    for (int tt = 0; tt < 256; ++tt) acc += sE[half * 256 + tt] * (float)vp[(size_t)tt * VSZ];
    sred[tid] = acc;
    __syncthreads();
    if (tid < VSZ) {
        const float cv = (sred[tid] + sred[tid + 128]) * inv;
        ctx[n * VSZ + tid] = cv;
        sctx[tid] = cv;
    }
    __syncthreads();

    if (tid < VOC) {
        const float* wr = w_out + tid * (KSZ + VSZ);
        float p = b_out[tid];
#pragma unroll 8
        for (int k = 0; k < KSZ; ++k) p += sh2[k] * wr[k];
#pragma unroll 8
        for (int k = 0; k < VSZ; ++k) p += sctx[k] * wr[KSZ + k];
        out[((size_t)n * LSTEPS + l) * VOC + tid] = p;
    }
}

// ---- host side ------------------------------------------------------------
extern "C" void kernel_launch(void* const* d_in, const int* in_sizes, int n_in,
                              void* d_out, int out_size, void* d_ws, size_t ws_size,
                              hipStream_t stream) {
    const float* enc_key = (const float*)d_in[0];
    const float* enc_val = (const float*)d_in[1];
    const int*   lens    = (const int*)d_in[2];
    const int*   text    = (const int*)d_in[3];
    const float* emb     = (const float*)d_in[4];
    const float* w_ih1   = (const float*)d_in[5];
    const float* w_hh1   = (const float*)d_in[6];
    const float* b_ih1   = (const float*)d_in[7];
    const float* b_hh1   = (const float*)d_in[8];
    const float* w_ih2   = (const float*)d_in[9];
    const float* w_hh2   = (const float*)d_in[10];
    const float* b_ih2   = (const float*)d_in[11];
    const float* b_hh2   = (const float*)d_in[12];
    const float* w_out   = (const float*)d_in[13];
    const float* b_out   = (const float*)d_in[14];
    float* out = (float*)d_out;

    char* p = (char*)d_ws;
    auto carve = [&](size_t bytes) { void* r = (void*)p; p += (bytes + 255) & ~(size_t)255; return r; };
    __bf16* W1cat = (__bf16*)carve((size_t)G1 * K1 * 2);
    __bf16* W2cat = (__bf16*)carve((size_t)G2 * K2 * 2);
    float*  b1    = (float*)carve(G1 * 4);
    float*  b2    = (float*)carve(G2 * 4);
    __bf16* keyT  = (__bf16*)carve((size_t)NB * T_ENC * KSZ * 2);
    __bf16* valT  = (__bf16*)carve((size_t)NB * T_ENC * VSZ * 2);
    float*  h1    = (float*)carve((size_t)NB * HSZ * 4);
    float*  c1    = (float*)carve((size_t)NB * HSZ * 4);
    float*  h2    = (float*)carve((size_t)NB * KSZ * 4);
    float*  c2    = (float*)carve((size_t)NB * KSZ * 4);
    float*  ctx   = (float*)carve((size_t)NB * VSZ * 4);
    __bf16* xcat  = (__bf16*)carve((size_t)NB * K1 * 2);
    __bf16* x2cat = (__bf16*)carve((size_t)NB * K2 * 2);
    float*  gates1 = (float*)carve((size_t)NB * G1 * 4);
    float*  gates2 = (float*)carve((size_t)NB * G2 * 4);

    prep_w1cat<<<(G1 * K1 + 255) / 256, 256, 0, stream>>>(w_ih1, w_hh1, W1cat);
    prep_w2cat<<<(G2 * K2 + 255) / 256, 256, 0, stream>>>(w_ih2, w_hh2, W2cat);
    prep_bias<<<(G1 + 255) / 256, 256, 0, stream>>>(b_ih1, b_hh1, b_ih2, b_hh2, b1, b2);
    transpose_enc<<<T_ENC * NB, 128, 0, stream>>>(enc_key, enc_val, keyT, valT);
    init_state<<<(NB * HSZ + 255) / 256, 256, 0, stream>>>(enc_val, h1, c1, h2, c2, ctx);

    for (int l = 0; l < LSTEPS; ++l) {
        build_x1<<<NB, 256, 0, stream>>>(text, emb, ctx, h1, xcat, l);
        // grid = (M/16/8) * (NOUT/32)
        gemm_bf16_wmma<NB, G1, K1><<<2 * (G1 / 32), 256, 0, stream>>>(xcat, W1cat, b1, gates1);
        lstm_act1<<<(NB * HSZ) / 256, 256, 0, stream>>>(gates1, c1, h1, h2, x2cat);
        gemm_bf16_wmma<NB, G2, K2><<<2 * (G2 / 32), 256, 0, stream>>>(x2cat, W2cat, b2, gates2);
        attn_fused<<<NB, 256, 0, stream>>>(gates2, c2, h2, keyT, valT, lens, w_out, b_out, ctx, out, l);
    }
}
